// TwoTowerBaseRetrieval_80659485819331
// MI455X (gfx1250) — matrix operand order repaired and checked by reading, not compiled, not run on hardware
//
#include <hip/hip_runtime.h>
#include <hip/hip_bf16.h>

typedef __attribute__((ext_vector_type(16))) __bf16   v16bf;
typedef __attribute__((ext_vector_type(8)))  __bf16   v8bf;
typedef __attribute__((ext_vector_type(8)))  float    v8f;
typedef __attribute__((ext_vector_type(4)))  float    v4f;
typedef __attribute__((ext_vector_type(4)))  unsigned u32x4;
typedef __attribute__((ext_vector_type(8)))  int      i32x8;
typedef __attribute__((ext_vector_type(4)))  int      i32x4;

#define NUSER   256
#define DU      128
#define IU      256
#define HLEN    200
#define NCAT    500000
#define TOPK    100
#define NWG     245                   // ceil(500000 / 2048)
#define ITERS   8                     // 8 * 256 items per WG = 2048
#define ITEMS_PER_WG (ITERS * 256)
#define NBKT    128                   // histogram buckets over score range [-0.5, 3.5)
#define CAP     2048                  // candidate capacity per user

// workspace layout (bytes)
#define WS_UBF    0                           // bf16 [256][128]  = 65536
#define WS_HIST   65536                       // u32  [256][128]  = 131072
#define WS_COUNT  (WS_HIST + 131072)          // u32  [256]       = 1024
#define WS_THR    (WS_COUNT + 1024)           // f32  [256]       = 1024
#define WS_CVAL   262144                      // f32  [256][CAP]  = 2 MB
#define WS_CIDX   (WS_CVAL + NUSER*CAP*4)     // i32  [256][CAP]  = 2 MB

// LDS sizes for the scoring kernel
#define LDS_UEMB   65536                      // bf16 [256][128]
#define LDS_HIST   65536                      // u32 [128 pairs][128 bkts] (2x u16 packed)

// ---------------------------------------------------------------- zero scratch
__global__ void zero_kernel(unsigned* __restrict__ p, int n) {
    int i = blockIdx.x * blockDim.x + threadIdx.x;
    if (i < n) p[i] = 0u;
}

// ---------------------------------------------------------------- user tower
__global__ void __launch_bounds__(128)
tower_kernel(const int* __restrict__ user_id,
             const float* __restrict__ user_features,
             const int* __restrict__ user_history,
             const float* __restrict__ user_id_emb,
             const float* __restrict__ item_id_emb,
             const float* __restrict__ W_feat,  const float* __restrict__ b_feat,
             const float* __restrict__ W_tower, const float* __restrict__ b_tower,
             __bf16* __restrict__ ubf)
{
    __shared__ float feats[IU];
    __shared__ int   hidx[HLEN];
    __shared__ float tin[3 * DU];

    const int u = blockIdx.x;
    const int d = threadIdx.x;

    feats[d]       = user_features[u * IU + d];
    feats[d + 128] = user_features[u * IU + d + 128];
    if (d < 100) {
        hidx[d]       = user_history[u * HLEN + d];
        hidx[d + 100] = user_history[u * HLEN + d + 100];
    }
    __syncthreads();

    float hs = 0.0f;
    for (int h = 0; h < HLEN; ++h)
        hs += item_id_emb[(size_t)hidx[h] * DU + d];
    hs *= (1.0f / (float)HLEN);

    float ue = user_id_emb[(size_t)user_id[u] * DU + d];

    float fa = b_feat[d];
    const float* wf = W_feat + (size_t)d * IU;
    for (int i = 0; i < IU; ++i) fa += wf[i] * feats[i];

    tin[d]          = ue;
    tin[DU + d]     = fa;
    tin[2 * DU + d] = hs;
    __syncthreads();

    float o = b_tower[d];
    const float* wt = W_tower + (size_t)d * (3 * DU);
    for (int i = 0; i < 3 * DU; ++i) o += wt[i] * tin[i];

    ubf[(size_t)u * DU + d] = (__bf16)o;
}

// ------------------------------------------------- TDM staging of user embeds
// Stage 64 KB bf16 user-embedding table into LDS offset 0. Uses the Tensor
// Data Mover (1-D D#: dim0 = tile0 = stride0 = 16384 dwords, data_size = 4B,
// type = 2) when the builtin exists; plain copy fallback otherwise.
__device__ __forceinline__ void stage_uemb(const __bf16* __restrict__ uemb_g,
                                           char* __restrict__ smem)
{
#if defined(__AMDGCN__) && __has_builtin(__builtin_amdgcn_tensor_load_to_lds) && \
    __has_builtin(__builtin_amdgcn_s_wait_tensorcnt)
    if (threadIdx.x < 32) {                         // one wave issues the DMA
        unsigned long long ga = (unsigned long long)(const void*)uemb_g;
        u32x4 g0;
        g0[0] = 1u;                                  // count=1 (valid descriptor)
        g0[1] = 0u;                                  // lds_addr = 0
        g0[2] = (unsigned)(ga & 0xffffffffu);        // global_addr[31:0]
        g0[3] = (unsigned)((ga >> 32) & 0x01ffffffu) // global_addr[56:32]
              | (2u << 30);                          // type = 2 ("image")
        i32x8 g1;
        g1[0] = (int)(2u << 16);                     // data_size = 4 bytes
        g1[1] = (int)(16384u << 16);                 // tensor_dim0[15:0] @ [63:48]
        g1[2] = 0;                                   // tensor_dim0[31:16], dim1 lo
        g1[3] = (int)(16384u << 16);                 // tile_dim0 = 16384 @ [127:112]
        g1[4] = 0;                                   // tile_dim1/2 unused
        g1[5] = 16384;                               // tensor_dim0_stride[31:0]
        g1[6] = 0;
        g1[7] = 0;
        i32x4 z4 = {0, 0, 0, 0};
#if defined(__clang_major__) && (__clang_major__ >= 23)
        i32x8 z8 = {0, 0, 0, 0, 0, 0, 0, 0};
        __builtin_amdgcn_tensor_load_to_lds(g0, g1, z4, z4, z8, 0);
#else
        __builtin_amdgcn_tensor_load_to_lds(g0, g1, z4, z4, 0);
#endif
        __builtin_amdgcn_s_wait_tensorcnt(0);
    }
#else
    const unsigned* src = (const unsigned*)uemb_g;
    unsigned* dst = (unsigned*)smem;
    for (int i = threadIdx.x; i < NUSER * DU / 2; i += 256) dst[i] = src[i];
#endif
}

// ---------------------------------------------------------------- WMMA scoring
// Each wave: 32 items (two 16-wide B tiles, all 4 K-steps live in VGPRs)
// x all 256 users (16 A tiles from LDS). Catalog read exactly once per pass.
// PHASE 0: packed u16x2 per-user score histogram in LDS -> global histogram
// PHASE 1: collect (score,idx) candidates with score >= per-user threshold
template <int PHASE>
__global__ void __launch_bounds__(256)
score_kernel(const float* __restrict__ catalog,
             const __bf16* __restrict__ uemb_g,
             unsigned* __restrict__ ghist,
             const float* __restrict__ gthr,
             unsigned* __restrict__ gcount,
             float* __restrict__ cval,
             int* __restrict__ cidx)
{
    extern __shared__ __align__(16) char smem[];
    const __bf16* uembLds = (const __bf16*)smem;        // 64 KB
    unsigned* histLds = (unsigned*)(smem + LDS_UEMB);   // phase0: 64 KB packed
    float*    thrLds  = (float*)(smem + LDS_UEMB);      // phase1: 1 KB

    const int slab = blockIdx.x * ITEMS_PER_WG;
    if (slab >= NCAT) return;                           // uniform early-exit

    stage_uemb(uemb_g, smem);
    if (PHASE == 0) {
        for (int i = threadIdx.x; i < 128 * NBKT; i += 256) histLds[i] = 0u;
    } else {
        thrLds[threadIdx.x] = gthr[threadIdx.x];
    }
    __syncthreads();

    // force wave id scalar so tile bounds become SGPR / s_cbranch control flow
    const int wave = __builtin_amdgcn_readfirstlane(threadIdx.x >> 5);
    const int lane = threadIdx.x & 31;

    const int nCol  = lane & 15;           // B column (item within 16-tile)
    const int kHalf = (lane >> 4) << 3;    // B K-select: 0 or 8
    const int aSel  = (lane >> 4) << 3;    // A K-select: 0 or 8

    for (int it = 0; it < ITERS; ++it) {
        const int tileBase = slab + it * 256 + wave * 32;   // wave-uniform scalar
        const bool tileFull = (tileBase + 32) <= NCAT;      // scalar fast-path flag
        const int item0 = tileBase + nCol;          // B tile 0
        const int item1 = tileBase + 16 + nCol;     // B tile 1
        const size_t c0 = (size_t)(item0 < NCAT ? item0 : NCAT - 1);
        const size_t c1 = (size_t)(item1 < NCAT ? item1 : NCAT - 1);
        const float* cp0 = catalog + c0 * DU + kHalf;
        const float* cp1 = catalog + c1 * DU + kHalf;

        v16bf bfrag0[4], bfrag1[4];
#pragma unroll
        for (int ks = 0; ks < 4; ++ks) {
            const v4f* pv0 = (const v4f*)(cp0 + ks * 32);
            const v4f* pv1 = (const v4f*)(cp1 + ks * 32);
            v4f a0 = pv0[0], a1 = pv0[1], a2 = pv0[4], a3 = pv0[5];
            v4f b0 = pv1[0], b1 = pv1[1], b2 = pv1[4], b3 = pv1[5];
            v16bf x, y;
#pragma unroll
            for (int e = 0; e < 4; ++e) {
                x[e] = (__bf16)a0[e]; x[4+e] = (__bf16)a1[e];
                x[8+e] = (__bf16)a2[e]; x[12+e] = (__bf16)a3[e];
                y[e] = (__bf16)b0[e]; y[4+e] = (__bf16)b1[e];
                y[8+e] = (__bf16)b2[e]; y[12+e] = (__bf16)b3[e];
            }
            bfrag0[ks] = x;
            bfrag1[ks] = y;
        }
        if (it + 1 < ITERS) {
            __builtin_prefetch(cp0 + 256 * DU, 0, 0);
            __builtin_prefetch(cp1 + 256 * DU, 0, 0);
        }

        for (int utile = 0; utile < 16; ++utile) {
            // preload all 4 A fragments -> one ds-wait, then 8 back-to-back WMMAs
            v16bf afrag[4];
            const __bf16* up = uembLds + (size_t)(utile * 16 + nCol) * DU + aSel;
#pragma unroll
            for (int ks = 0; ks < 4; ++ks) {
                v8bf lo = *(const v8bf*)(up + ks * 32);
                v8bf hi = *(const v8bf*)(up + ks * 32 + 16);
                v16bf a;
#pragma unroll
                for (int e = 0; e < 8; ++e) { a[e] = lo[e]; a[8+e] = hi[e]; }
                afrag[ks] = a;
            }
            v8f acc0 = {0.f,0.f,0.f,0.f,0.f,0.f,0.f,0.f};
            v8f acc1 = {0.f,0.f,0.f,0.f,0.f,0.f,0.f,0.f};
#pragma unroll
            for (int ks = 0; ks < 4; ++ks) {
                acc0 = __builtin_amdgcn_wmma_f32_16x16x32_bf16(
                           false, afrag[ks], false, bfrag0[ks], (short)0, acc0, false, false);
                acc1 = __builtin_amdgcn_wmma_f32_16x16x32_bf16(
                           false, afrag[ks], false, bfrag1[ks], (short)0, acc1, false, false);
            }

            // acc[r]: user = utile*16 + r + (lane>=16 ? 8 : 0)
            const int userBase = utile * 16 + ((lane >> 4) << 3);

            if (PHASE == 0) {
                if (tileFull) {                       // scalar branch: no exec masking
#pragma unroll
                    for (int r = 0; r < 8; ++r) {
                        const int user = userBase + r;
                        const unsigned inc = (user & 1) ? 65536u : 1u;
                        int b0 = (int)(acc0[r] * 32.0f + 16.0f);
                        b0 = b0 < 0 ? 0 : (b0 > NBKT - 1 ? NBKT - 1 : b0);
                        atomicAdd(&histLds[(user >> 1) * NBKT + b0], inc);
                        int b1 = (int)(acc1[r] * 32.0f + 16.0f);
                        b1 = b1 < 0 ? 0 : (b1 > NBKT - 1 ? NBKT - 1 : b1);
                        atomicAdd(&histLds[(user >> 1) * NBKT + b1], inc);
                    }
                } else {                              // boundary tile only
                    const bool valid0 = item0 < NCAT;
                    const bool valid1 = item1 < NCAT;
#pragma unroll
                    for (int r = 0; r < 8; ++r) {
                        const int user = userBase + r;
                        const unsigned inc = (user & 1) ? 65536u : 1u;
                        if (valid0) {
                            int b0 = (int)(acc0[r] * 32.0f + 16.0f);
                            b0 = b0 < 0 ? 0 : (b0 > NBKT - 1 ? NBKT - 1 : b0);
                            atomicAdd(&histLds[(user >> 1) * NBKT + b0], inc);
                        }
                        if (valid1) {
                            int b1 = (int)(acc1[r] * 32.0f + 16.0f);
                            b1 = b1 < 0 ? 0 : (b1 > NBKT - 1 ? NBKT - 1 : b1);
                            atomicAdd(&histLds[(user >> 1) * NBKT + b1], inc);
                        }
                    }
                }
            } else {
                if (tileFull) {
#pragma unroll
                    for (int r = 0; r < 8; ++r) {
                        const int user = userBase + r;
                        const float thr = thrLds[user];
                        if (acc0[r] >= thr) {
                            unsigned pos = atomicAdd(&gcount[user], 1u);
                            if (pos < CAP) {
                                cval[(size_t)user * CAP + pos] = acc0[r];
                                cidx[(size_t)user * CAP + pos] = item0;
                            }
                        }
                        if (acc1[r] >= thr) {
                            unsigned pos = atomicAdd(&gcount[user], 1u);
                            if (pos < CAP) {
                                cval[(size_t)user * CAP + pos] = acc1[r];
                                cidx[(size_t)user * CAP + pos] = item1;
                            }
                        }
                    }
                } else {
                    const bool valid0 = item0 < NCAT;
                    const bool valid1 = item1 < NCAT;
#pragma unroll
                    for (int r = 0; r < 8; ++r) {
                        const int user = userBase + r;
                        const float thr = thrLds[user];
                        if (valid0 && acc0[r] >= thr) {
                            unsigned pos = atomicAdd(&gcount[user], 1u);
                            if (pos < CAP) {
                                cval[(size_t)user * CAP + pos] = acc0[r];
                                cidx[(size_t)user * CAP + pos] = item0;
                            }
                        }
                        if (valid1 && acc1[r] >= thr) {
                            unsigned pos = atomicAdd(&gcount[user], 1u);
                            if (pos < CAP) {
                                cval[(size_t)user * CAP + pos] = acc1[r];
                                cidx[(size_t)user * CAP + pos] = item1;
                            }
                        }
                    }
                }
            }
        }
    }

    if (PHASE == 0) {
        __syncthreads();
        // flush packed LDS histogram (128 pairs x 128 buckets) to global u32 hist
        for (int i = threadIdx.x; i < 128 * NBKT; i += 256) {
            const unsigned w = histLds[i];
            if (w) {
                const int pair = i / NBKT;
                const int bkt  = i - pair * NBKT;
                const unsigned lo = w & 0xffffu;
                const unsigned hi = w >> 16;
                if (lo) atomicAdd(&ghist[(2 * pair)     * NBKT + bkt], lo);
                if (hi) atomicAdd(&ghist[(2 * pair + 1) * NBKT + bkt], hi);
            }
        }
    }
}

// ---------------------------------------------------------------- thresholds
__global__ void thresh_kernel(const unsigned* __restrict__ ghist,
                              float* __restrict__ gthr)
{
    const int u = threadIdx.x;
    const unsigned* h = ghist + u * NBKT;
    unsigned cum = 0;
    float thr = -1.0e30f;
    for (int b = NBKT - 1; b >= 0; --b) {
        cum += h[b];
        if (cum >= TOPK) { thr = (float)(b - 16) * (1.0f / 32.0f); break; }
    }
    gthr[u] = thr;
}

// ---------------------------------------------------------------- final select
__global__ void __launch_bounds__(256)
select_kernel(const unsigned* __restrict__ gcount,
              const float* __restrict__ cval,
              const int* __restrict__ cidx,
              float* __restrict__ out)
{
    __shared__ float v[CAP];
    __shared__ int  ix[CAP];
    const int u = blockIdx.x;
    unsigned c = gcount[u];
    const int n = (c > (unsigned)CAP) ? CAP : (int)c;

    for (int i = threadIdx.x; i < CAP; i += 256) {
        if (i < n) { v[i] = cval[(size_t)u * CAP + i]; ix[i] = cidx[(size_t)u * CAP + i]; }
        else       { v[i] = -3.0e38f;                  ix[i] = 0x7fffffff; }
    }
    __syncthreads();

    for (int k = 2; k <= CAP; k <<= 1) {
        for (int j = k >> 1; j > 0; j >>= 1) {
            for (int i = threadIdx.x; i < CAP; i += 256) {
                const int p = i ^ j;
                if (p > i) {
                    const bool up = ((i & k) == 0);
                    float va = v[i], vb = v[p];
                    int   ia = ix[i], ib = ix[p];
                    const bool aFirst = (va > vb) || (va == vb && ia < ib);
                    if (up != aFirst) { v[i] = vb; v[p] = va; ix[i] = ib; ix[p] = ia; }
                }
            }
            __syncthreads();
        }
    }

    if (threadIdx.x < TOPK) {
        out[(size_t)u * TOPK + threadIdx.x]                        = v[threadIdx.x];
        out[(size_t)NUSER * TOPK + (size_t)u * TOPK + threadIdx.x] = (float)ix[threadIdx.x];
    }
}

// ---------------------------------------------------------------- launch
extern "C" void kernel_launch(void* const* d_in, const int* in_sizes, int n_in,
                              void* d_out, int out_size, void* d_ws, size_t ws_size,
                              hipStream_t stream)
{
    const int*   user_id       = (const int*)d_in[0];
    const float* user_features = (const float*)d_in[1];
    const int*   user_history  = (const int*)d_in[2];
    const float* user_id_emb   = (const float*)d_in[3];
    const float* item_id_emb   = (const float*)d_in[4];
    const float* W_feat        = (const float*)d_in[5];
    const float* b_feat        = (const float*)d_in[6];
    const float* W_tower       = (const float*)d_in[7];
    const float* b_tower       = (const float*)d_in[8];
    const float* catalog       = (const float*)d_in[9];
    (void)in_sizes; (void)n_in; (void)out_size; (void)ws_size;

    char* ws = (char*)d_ws;
    __bf16*   ubf    = (__bf16*)(ws + WS_UBF);
    unsigned* ghist  = (unsigned*)(ws + WS_HIST);
    unsigned* gcount = (unsigned*)(ws + WS_COUNT);
    float*    gthr   = (float*)(ws + WS_THR);
    float*    cvalp  = (float*)(ws + WS_CVAL);
    int*      cidxp  = (int*)(ws + WS_CIDX);

    hipFuncSetAttribute((const void*)score_kernel<0>,
                        hipFuncAttributeMaxDynamicSharedMemorySize, LDS_UEMB + LDS_HIST);
    hipFuncSetAttribute((const void*)score_kernel<1>,
                        hipFuncAttributeMaxDynamicSharedMemorySize, LDS_UEMB + 1024);

    // 1) zero histogram + counters (ws is poisoned by harness)
    zero_kernel<<<(NUSER * NBKT + NUSER + 255) / 256, 256, 0, stream>>>(ghist, NUSER * NBKT + NUSER);

    // 2) user tower -> bf16 user embeddings
    tower_kernel<<<NUSER, 128, 0, stream>>>(user_id, user_features, user_history,
                                            user_id_emb, item_id_emb,
                                            W_feat, b_feat, W_tower, b_tower, ubf);

    // 3) WMMA scoring pass 0: per-user score histogram
    score_kernel<0><<<NWG, 256, LDS_UEMB + LDS_HIST, stream>>>(
        catalog, ubf, ghist, gthr, gcount, cvalp, cidxp);

    // 4) per-user threshold from histogram
    thresh_kernel<<<1, 256, 0, stream>>>(ghist, gthr);

    // 5) WMMA scoring pass 1: collect candidates above threshold
    score_kernel<1><<<NWG, 256, LDS_UEMB + 1024, stream>>>(
        catalog, ubf, ghist, gthr, gcount, cvalp, cidxp);

    // 6) per-user bitonic top-100 (vals then indices, lax.top_k tie order)
    select_kernel<<<NUSER, 256, 0, stream>>>(gcount, cvalp, cidxp, (float*)d_out);
}
